// MaceCoreModel_37434934952448
// MI455X (gfx1250) — matrix operand order repaired and checked by reading, not compiled, not run on hardware
//
#include <hip/hip_runtime.h>
#include <hip/hip_bf16.h>
#include <math.h>

#define NNODES 8192
#define NEDGES 131072
#define ZSP    10
#define CCH    128
#define NBASIS 8
#define NCW    ((size_t)NNODES * CCH)       // 1,048,576
#define NC16   ((size_t)16 * NCW)           // 16,777,216

typedef __attribute__((ext_vector_type(16))) __bf16 v16bf;
typedef __attribute__((ext_vector_type(8)))  float  v8f;

struct bf16x4 { __bf16 x, y, z, w; };       // 8-byte packed LDS store

__device__ __forceinline__ float silu_f(float x) { return x / (1.0f + __expf(-x)); }

// ---------------------------------------------------------------------------
// bf16 WMMA GEMM: D[M x Ncols] = act( A[M x K] @ B[K x Ncols] + Add )
// M % 64 == 0, K % 32 == 0, Ncols % BN == 0. 256 threads = 8 wave32.
// Block tile 64 x BN; each wave owns 16x(16*TPW) of output -> TPW wmma / K-step.
// K-tiles are software-pipelined through registers: global loads for tile k+1
// are issued before the WMMA phase of tile k, so vmem latency hides under math.
// A is stored K-swizzled in LDS so every fragment read is 32 contiguous bytes.
// ---------------------------------------------------------------------------
template <int BN>
__global__ __launch_bounds__(256) void gemm_bf16_wmma(
    const float* __restrict__ A, const float* __restrict__ B,
    const float* __restrict__ Add, float* __restrict__ D,
    int M, int K, int Ncols, int act)
{
  constexpr int TPW = BN / 32;              // col tiles per wave (4 or 2)
  constexpr int NB4 = (32 * BN / 4) / 256;  // B float4 chunks per thread (4 or 2)
  __shared__ __bf16 As[64][48];             // 64x32 used, row = 96B (32B-aligned)
  __shared__ __bf16 Bs[32][BN + 16];        // row = (BN+16)*2 B (32B-aligned)

  const int tid   = threadIdx.x;
  const int lane  = tid & 31;
  const int wave  = tid >> 5;               // 0..7
  const int trow  = wave & 3;               // row tile 0..3
  const int tcol0 = (wave >> 2) * TPW;      // first col tile for this wave

  const size_t rowBase = (size_t)blockIdx.y * 64;
  const size_t colBase = (size_t)blockIdx.x * BN;

  // ---- fixed per-thread staging coordinates ----
  const int rA0 = tid >> 3;                 // A chunk 0: row, col*4
  const int cA0 = (tid & 7) * 4;
  const int rA1 = rA0 + 32;                 // A chunk 1 (tid+256), same column
  const int dA  = (cA0 >= 8 && cA0 < 24) ? (cA0 ^ 24) : cA0;  // K swizzle
  const float* pA0 = A + (rowBase + rA0) * (size_t)K + cA0;
  const float* pA1 = A + (rowBase + rA1) * (size_t)K + cA0;
  int rB[NB4], cB[NB4];
  const float* pB[NB4];
#pragma unroll
  for (int j = 0; j < NB4; ++j) {
    int ch = tid + j * 256;
    rB[j] = ch / (BN / 4);
    cB[j] = (ch % (BN / 4)) * 4;
    pB[j] = B + (size_t)rB[j] * Ncols + colBase + cB[j];
  }

  // ---- prologue: load K-tile 0 into registers ----
  float4 a0 = *(const float4*)pA0;
  float4 a1 = *(const float4*)pA1;
  float4 bb[NB4];
#pragma unroll
  for (int j = 0; j < NB4; ++j) bb[j] = *(const float4*)pB[j];

  v8f acc[TPW];
#pragma unroll
  for (int t = 0; t < TPW; ++t) acc[t] = (v8f){};

  const int half = lane >> 4;
  const int mloc = trow * 16 + (lane & 15);
  const int kB   = half << 3;                   // B fragment: k base, table order

  for (int kk = 0; kk < K; kk += 32) {
    __syncthreads();   // previous iteration's fragment reads complete
    // ---- convert + store staged registers to LDS ----
    *(bf16x4*)&As[rA0][dA] = {(__bf16)a0.x, (__bf16)a0.y, (__bf16)a0.z, (__bf16)a0.w};
    *(bf16x4*)&As[rA1][dA] = {(__bf16)a1.x, (__bf16)a1.y, (__bf16)a1.z, (__bf16)a1.w};
#pragma unroll
    for (int j = 0; j < NB4; ++j)
      *(bf16x4*)&Bs[rB[j]][cB[j]] =
          {(__bf16)bb[j].x, (__bf16)bb[j].y, (__bf16)bb[j].z, (__bf16)bb[j].w};
    __syncthreads();
    // ---- issue next K-tile's global loads (in flight during WMMA) ----
    if (kk + 32 < K) {
      pA0 += 32; pA1 += 32;
      a0 = *(const float4*)pA0;
      a1 = *(const float4*)pA1;
#pragma unroll
      for (int j = 0; j < NB4; ++j) {
        pB[j] += (size_t)32 * Ncols;
        bb[j] = *(const float4*)pB[j];
      }
    }
    // ---- fragments (A: 2x ds_load_b128; B: 16 strided 2-byte reads) + WMMA ----
    v16bf af = *(const v16bf*)&As[mloc][half * 16];   // K-swizzle gives ISA order
#pragma unroll
    for (int t = 0; t < TPW; ++t) {
      v16bf bfv;
#pragma unroll
      for (int s = 0; s < 16; ++s)
        bfv[s] = Bs[kB + (s & 7) + ((s >> 3) << 4)][(tcol0 + t) * 16 + (lane & 15)];
      acc[t] = __builtin_amdgcn_wmma_f32_16x16x32_bf16(false, af, false, bfv,
                                                       (short)0, acc[t], false, false);
    }
  }

  // ---- epilogue: C/D layout  n = lane&15, m = v + 8*(lane>>4) ----
  const int mbase = trow * 16 + (lane >> 4) * 8;
#pragma unroll
  for (int t = 0; t < TPW; ++t) {
    int nn = (tcol0 + t) * 16 + (lane & 15);
#pragma unroll
    for (int v = 0; v < 8; ++v) {
      size_t row = rowBase + mbase + v;
      size_t idx = row * (size_t)Ncols + colBase + nn;
      float r0 = acc[t][v];
      if (Add) r0 += Add[idx];
      if (act) r0 = silu_f(r0);
      D[idx] = r0;
    }
  }
}

// ---------------------------------------------------------------------------
// Edge geometry: unit vector -> real SH (l<=3, component norm) + Bessel basis
// ---------------------------------------------------------------------------
__global__ __launch_bounds__(256) void edge_geom_kernel(
    const float* __restrict__ pos, const float* __restrict__ shifts,
    const int* __restrict__ ei, float* __restrict__ sh, float* __restrict__ ef)
{
  int e = blockIdx.x * 256 + threadIdx.x;
  if (e >= NEDGES) return;
  int s = ei[e], r = ei[NEDGES + e];
  float vx = pos[r * 3 + 0] - pos[s * 3 + 0] + shifts[e * 3 + 0];
  float vy = pos[r * 3 + 1] - pos[s * 3 + 1] + shifts[e * 3 + 1];
  float vz = pos[r * 3 + 2] - pos[s * 3 + 2] + shifts[e * 3 + 2];
  float len = sqrtf(vx * vx + vy * vy + vz * vz) + 1e-9f;
  float x = vx / len, y = vy / len, z = vz / len;
  float x2 = x * x, y2 = y * y, z2 = z * z;
  const float s3 = 1.7320508075688772f, s5 = 2.23606797749979f, s15 = 3.872983346207417f;
  const float a33 = 2.091650066335189f;   // 0.25*sqrt(70)
  const float a32 = 10.246950765959598f;  // sqrt(105)
  const float a31 = 1.6201851746019651f;  // 0.25*sqrt(42)
  const float a30 = 1.3228756555322954f;  // 0.5*sqrt(7)
  float* o = sh + (size_t)e * 16;
  o[0]  = 1.0f;
  o[1]  = s3 * x;  o[2] = s3 * y;  o[3] = s3 * z;
  o[4]  = s15 * x * y;
  o[5]  = s15 * y * z;
  o[6]  = 0.5f * s5 * (3.0f * z2 - 1.0f);
  o[7]  = s15 * x * z;
  o[8]  = 0.5f * s15 * (x2 - y2);
  o[9]  = a33 * y * (3.0f * x2 - y2);
  o[10] = a32 * x * y * z;
  o[11] = a31 * y * (5.0f * z2 - 1.0f);
  o[12] = a30 * z * (5.0f * z2 - 3.0f);
  o[13] = a31 * x * (5.0f * z2 - 1.0f);
  o[14] = 0.5f * a32 * z * (x2 - y2);
  o[15] = a33 * x * (x2 - 3.0f * y2);

  float xr = len * 0.2f;   // r / R_MAX
  float cut = 0.0f;
  if (xr < 1.0f) {
    float x5 = xr * xr * xr * xr * xr;
    cut = 1.0f - 21.0f * x5 + 35.0f * x5 * xr - 15.0f * x5 * xr * xr;  // p=5 cutoff
  }
  const float pref = 0.6324555320336759f;  // sqrt(2/R_MAX)
  float* fo = ef + (size_t)e * 8;
#pragma unroll
  for (int n = 1; n <= 8; ++n)
    fo[n - 1] = pref * __sinf((float)n * 3.14159265358979323846f * xr) / len * cut;
}

// feats[m][n][c]: m=0 <- node_attrs @ W_embed (one-hot matvec), others 0
__global__ __launch_bounds__(256) void init_feats_kernel(
    const float* __restrict__ na, const float* __restrict__ We, float* __restrict__ feats)
{
  size_t idx = (size_t)blockIdx.x * 256 + threadIdx.x;
  size_t m = idx / NCW, rem = idx - m * NCW;
  float v = 0.0f;
  if (m == 0) {
    int n = (int)(rem >> 7), c = (int)(rem & 127);
#pragma unroll
    for (int z = 0; z < ZSP; ++z) v += na[n * ZSP + z] * We[z * CCH + c];
  }
  feats[idx] = v;
}

// r1 = silu(edge_feats @ R_w1[L])  (K=8 too small for WMMA)
__global__ __launch_bounds__(256) void radial1_kernel(
    const float* __restrict__ ef, const float* __restrict__ W1, float* __restrict__ r1)
{
  int idx = blockIdx.x * 256 + threadIdx.x;   // E*64
  int e = idx >> 6, j = idx & 63;
  const float* f = ef + (size_t)e * 8;
  float acc = 0.0f;
#pragma unroll
  for (int b = 0; b < NBASIS; ++b) acc += f[b] * W1[b * 64 + j];
  r1[(size_t)e * 64 + j] = silu_f(acc);
}

__global__ __launch_bounds__(256) void zero_kernel(float* __restrict__ p, size_t n) {
  size_t i = (size_t)blockIdx.x * 256 + threadIdx.x;
  if (i < n) p[i] = 0.0f;
}

// Fused message + scatter: agg[m][recv][c] += h0[send][c]*w[e][c][lof(m)]*sh[e][m]/16
__global__ __launch_bounds__(256) void message_kernel(
    const float* __restrict__ h0, const float* __restrict__ wbuf,
    const float* __restrict__ sh, const int* __restrict__ ei, float* __restrict__ agg)
{
  const int e = blockIdx.x * 2 + (threadIdx.x >> 7);
  const int c = threadIdx.x & 127;
  const int snd = ei[e], rcv = ei[NEDGES + e];
  const float hs = h0[(size_t)snd * CCH + c];
  const float4 wv = *(const float4*)(wbuf + (size_t)e * 512 + c * 4);
  const float wl[4] = {wv.x, wv.y, wv.z, wv.w};
  const float4* shv = (const float4*)(sh + (size_t)e * 16);
  float she[16];
#pragma unroll
  for (int q = 0; q < 4; ++q) {
    float4 t = shv[q];
    she[q * 4 + 0] = t.x; she[q * 4 + 1] = t.y;
    she[q * 4 + 2] = t.z; she[q * 4 + 3] = t.w;
  }
  constexpr int LOFM[16] = {0,1,1,1,2,2,2,2,2,3,3,3,3,3,3,3};
#pragma unroll
  for (int m = 0; m < 16; ++m) {
    float val = hs * wl[LOFM[m]] * she[m] * (1.0f / 16.0f);
    atomicAdd(agg + (size_t)m * NCW + (size_t)rcv * CCH + c, val);
  }
}

// scale = node_attrs@w_prod . [1,s,s^2]; tmp = m*scale; sc = h*(node_attrs@w_sc)
__global__ __launch_bounds__(256) void scale_kernel(
    const float* __restrict__ na, const float* __restrict__ wprod,
    const float* __restrict__ wsc, const float* __restrict__ mbuf,
    const float* __restrict__ hbuf, float* __restrict__ tmp, float* __restrict__ scb)
{
  int idx = blockIdx.x * 256 + threadIdx.x;  // N*C
  int n = idx >> 7, c = idx & 127;
  float c0 = 0.f, c1 = 0.f, c2 = 0.f, sw = 0.f;
#pragma unroll
  for (int z = 0; z < ZSP; ++z) {
    float a = na[n * ZSP + z];
    const float* wp = wprod + ((size_t)z * CCH + c) * 3;
    c0 += a * wp[0]; c1 += a * wp[1]; c2 += a * wp[2];
    sw += a * wsc[z * CCH + c];
  }
  float s = mbuf[idx];                 // m = 0 slice
  float scale = c0 + c1 * s + c2 * s * s;
#pragma unroll
  for (int m = 0; m < 16; ++m) {
    size_t o = (size_t)m * NCW + idx;
    tmp[o] = mbuf[o] * scale;
    scb[o] = hbuf[o] * sw;
  }
}

__global__ __launch_bounds__(256) void readout0_kernel(
    const float* __restrict__ feats, const float* __restrict__ w, float* __restrict__ out)
{
  int n = blockIdx.x * 256 + threadIdx.x;
  if (n >= NNODES) return;
  const float* f = feats + (size_t)n * CCH;
  float acc = 0.f;
#pragma unroll 8
  for (int c = 0; c < CCH; ++c) acc += f[c] * w[c];
  out[n] = acc;
}

__global__ __launch_bounds__(256) void readout1_kernel(
    const float* __restrict__ feats, const float* __restrict__ w1,
    const float* __restrict__ w2, float* __restrict__ out)
{
  int n = blockIdx.x * 256 + threadIdx.x;
  if (n >= NNODES) return;
  const float* f = feats + (size_t)n * CCH;
  float hid[16];
#pragma unroll
  for (int j = 0; j < 16; ++j) hid[j] = 0.f;
  for (int c = 0; c < CCH; ++c) {
    float fv = f[c];
#pragma unroll
    for (int j = 0; j < 16; ++j) hid[j] += fv * w1[c * 16 + j];
  }
  float acc = 0.f;
#pragma unroll
  for (int j = 0; j < 16; ++j) acc += silu_f(hid[j]) * w2[j];
  out[n] += acc;
}

// ---------------------------------------------------------------------------
extern "C" void kernel_launch(void* const* d_in, const int* in_sizes, int n_in,
                              void* d_out, int out_size, void* d_ws, size_t ws_size,
                              hipStream_t stream)
{
  (void)in_sizes; (void)n_in; (void)out_size; (void)ws_size;
  const float* positions  = (const float*)d_in[0];
  const float* node_attrs = (const float*)d_in[1];
  const float* shifts     = (const float*)d_in[2];
  const float* W_embed    = (const float*)d_in[3];
  const float* W_up       = (const float*)d_in[4];
  const float* R_w1       = (const float*)d_in[5];
  const float* R_w2       = (const float*)d_in[6];
  const float* R_w3       = (const float*)d_in[7];
  const float* W_out      = (const float*)d_in[8];
  const float* w_sc       = (const float*)d_in[9];
  const float* w_prod     = (const float*)d_in[10];
  const float* W_plin     = (const float*)d_in[11];
  const float* w_ro0      = (const float*)d_in[12];
  const float* ro_w1      = (const float*)d_in[13];
  const float* ro_w2      = (const float*)d_in[14];
  const int*   eidx       = (const int*)d_in[15];
  float* out = (float*)d_out;

  float* ws = (float*)d_ws;
  size_t off = 0;
  float* sh    = ws + off; off += (size_t)NEDGES * 16;
  float* ef    = ws + off; off += (size_t)NEDGES * 8;
  float* r1    = ws + off; off += (size_t)NEDGES * 64;
  float* r2    = ws + off; off += (size_t)NEDGES * 64;
  float* wbuf  = ws + off; off += (size_t)NEDGES * 512;
  float* feats = ws + off; off += NC16;
  float* hbuf  = ws + off; off += NC16;
  float* agg   = ws + off; off += NC16;   // reused as `tmp` after W_out GEMM
  float* mbuf  = ws + off; off += NC16;
  float* scb   = ws + off; off += NC16;

  edge_geom_kernel<<<NEDGES / 256, 256, 0, stream>>>(positions, shifts, eidx, sh, ef);
  init_feats_kernel<<<(unsigned)(NC16 / 256), 256, 0, stream>>>(node_attrs, W_embed, feats);

  const int M16 = 16 * NNODES;  // 131072 rows for all m-major node GEMMs
  for (int L = 0; L < 2; ++L) {
    // radial MLP
    radial1_kernel<<<(NEDGES * 64) / 256, 256, 0, stream>>>(ef, R_w1 + L * NBASIS * 64, r1);
    gemm_bf16_wmma<64><<<dim3(1, NEDGES / 64), 256, 0, stream>>>(
        r1, R_w2 + L * 64 * 64, nullptr, r2, NEDGES, 64, 64, 1);
    gemm_bf16_wmma<128><<<dim3(4, NEDGES / 64), 256, 0, stream>>>(
        r2, R_w3 + (size_t)L * 64 * 512, nullptr, wbuf, NEDGES, 64, 512, 0);
    // h = feats @ W_up  (all 16 m slices as one GEMM)
    gemm_bf16_wmma<128><<<dim3(1, M16 / 64), 256, 0, stream>>>(
        feats, W_up + L * CCH * CCH, nullptr, hbuf, M16, CCH, CCH, 0);
    // message passing
    zero_kernel<<<(unsigned)(NC16 / 256), 256, 0, stream>>>(agg, NC16);
    message_kernel<<<NEDGES / 2, 256, 0, stream>>>(hbuf, wbuf, sh, eidx, agg);
    // m = agg @ W_out
    gemm_bf16_wmma<128><<<dim3(1, M16 / 64), 256, 0, stream>>>(
        agg, W_out + L * CCH * CCH, nullptr, mbuf, M16, CCH, CCH, 0);
    // nonlinearity scale + residual prep (tmp reuses agg)
    scale_kernel<<<(unsigned)(NCW / 256), 256, 0, stream>>>(
        node_attrs, w_prod + (size_t)L * ZSP * CCH * 3, w_sc + (size_t)L * ZSP * CCH,
        mbuf, hbuf, agg, scb);
    // feats = tmp @ W_plin + sc
    gemm_bf16_wmma<128><<<dim3(1, M16 / 64), 256, 0, stream>>>(
        agg, W_plin + L * CCH * CCH, scb, feats, M16, CCH, CCH, 0);
    // readout
    if (L == 0)
      readout0_kernel<<<NNODES / 256, 256, 0, stream>>>(feats, w_ro0, out);
    else
      readout1_kernel<<<NNODES / 256, 256, 0, stream>>>(feats, ro_w1, ro_w2, out);
  }
}
